// LayeredBayesianGraphLSTM_14310831030274
// MI455X (gfx1250) — matrix operand order repaired and checked by reading, not compiled, not run on hardware
//
#include <hip/hip_runtime.h>
#include <hip/hip_bf16.h>
#include <stdint.h>

typedef unsigned short u16;
typedef __attribute__((ext_vector_type(16))) __bf16 v16bf;
typedef __attribute__((ext_vector_type(8)))  float  v8f;

#define DEV static __device__ __forceinline__

// ---------------- problem constants ----------------
constexpr int S_ = 8, Bt = 256, H_ = 256, T_ = 90, FH_ = 14, FU_ = 76;
constexpr int G4H = 4 * H_;                 // 1024 gates
constexpr int CB = 64;                      // batch rows per workgroup
constexpr int CHUNKS = Bt / CB;             // 4
constexpr int NWG = S_ * CHUNKS;            // 32 workgroups
constexpr int THREADS = 1024;               // 32 wave32 (8 waves / SIMD32)
constexpr int NWAVE = THREADS / 32;
constexpr int PATCH = (CB / 16) * (H_ / 16) / NWAVE;  // 2 (16x16 tiles per wave)
constexpr int PARTS = THREADS / CB;         // 16 partial lanes per batch row in mix stage

// ---------------- workspace layout (bytes) ----------------
constexpr size_t B_BAR = 0;                                   // barrier slots (memset each call)
constexpr size_t B_XE  = 4096;
constexpr size_t SZ_XE = (size_t)S_ * T_  * Bt * 32 * 2;      // bf16, in padded to 32
constexpr size_t B_XF  = B_XE + SZ_XE;
constexpr size_t SZ_XF = (size_t)S_ * FH_ * Bt * 32 * 2;
constexpr size_t B_XU  = B_XF + SZ_XF;
constexpr size_t SZ_XU = (size_t)S_ * FU_ * Bt * 32 * 2;
constexpr size_t B_W   = B_XU + SZ_XU;
// per prefix (enc/df/du): Wih0 [S][1024][32] + Whh0/Wih1/Whh1 [S][1024][256], bf16
constexpr size_t O_WIH0 = 0;
constexpr size_t O_WHH0 = (size_t)S_ * G4H * 32;              // ushort offsets
constexpr size_t O_WIH1 = O_WHH0 + (size_t)S_ * G4H * H_;
constexpr size_t O_WHH1 = O_WIH1 + (size_t)S_ * G4H * H_;
constexpr size_t PFX_USH = O_WHH1 + (size_t)S_ * G4H * H_;    // ushorts per prefix
constexpr size_t B_WIN = B_W + 3 * PFX_USH * 2;
constexpr size_t SZ_WIN = (size_t)S_ * H_ * 288 * 2;          // K=257 padded to 288
constexpr size_t B_GWT = B_WIN + SZ_WIN;
constexpr size_t SZ_GWT = (size_t)H_ * H_ * 2;                // gW transposed, bf16
constexpr size_t B_XW  = B_GWT + SZ_GWT;                      // fp32, double buffered
constexpr size_t SZ_XW = 2ull * S_ * Bt * H_ * 4;

// ---------------- small helpers ----------------
DEV u16 f2bf(float f) {
  __bf16 b = (__bf16)f;
  return __builtin_bit_cast(unsigned short, b);
}
DEV float bf2f(u16 x) { return (float)__builtin_bit_cast(__bf16, x); }
DEV float sigmoidf_(float x) { return 1.0f / (1.0f + __expf(-x)); }
DEV float tanhf_(float x) { return 2.0f / (1.0f + __expf(-2.0f * x)) - 1.0f; }

// A-fragment: 16x32 bf16, row-major source, row = lane&15, halves K[kb..kb+7] / K[kb+16..kb+23]
DEV v16bf load_a(const u16* p, int ld, int lane) {
  const u16* r = p + (size_t)(lane & 15) * ld + ((lane >> 4) << 3);
  union { uint4 q[2]; v16bf v; } u;
  u.q[0] = *(const uint4*)(r);
  u.q[1] = *(const uint4*)(r + 16);
  return u.v;
}
// B-fragment: 32x16 bf16 with W stored [N][K] row-major: col = lane&15, halves K[kb..kb+7] / K[kb+16..kb+23]
DEV v16bf load_b(const u16* p, int ld, int lane) {
  const u16* r = p + (size_t)(lane & 15) * ld + ((lane >> 4) << 3);
  union { uint4 q[2]; v16bf v; } u;
  u.q[0] = *(const uint4*)(r);
  u.q[1] = *(const uint4*)(r + 16);
  return u.v;
}
DEV v8f wmma_bf(v16bf a, v16bf b, v8f c) {
  return __builtin_amdgcn_wmma_f32_16x16x32_bf16(false, a, false, b, (short)0, c, false, false);
}

// ---------------- conversion kernels ----------------
__global__ void cvt_pad(const float* __restrict__ src, u16* __restrict__ dst,
                        int R, int Cin, int Cpad) {
  size_t idx = (size_t)blockIdx.x * blockDim.x + threadIdx.x;
  if (idx >= (size_t)R * Cpad) return;
  int c = (int)(idx % Cpad);
  size_t r = idx / Cpad;
  float v = (c < Cin) ? src[r * Cin + c] : 0.0f;
  dst[idx] = f2bf(v);
}

// x [S,B,Td,Cin] fp32 -> [S,Td,B,32] bf16 (zero padded)
__global__ void cvt_x(const float* __restrict__ src, u16* __restrict__ dst, int Td, int Cin) {
  size_t idx = (size_t)blockIdx.x * blockDim.x + threadIdx.x;
  size_t N = (size_t)S_ * Td * Bt * 32;
  if (idx >= N) return;
  int k = (int)(idx & 31);
  size_t t1 = idx >> 5;
  int b = (int)(t1 % Bt); t1 /= Bt;
  int t = (int)(t1 % Td);
  int s = (int)(t1 / Td);
  float v = (k < Cin) ? src[(((size_t)s * Bt + b) * Td + t) * Cin + k] : 0.0f;
  dst[idx] = f2bf(v);
}

// gW [H,H] -> gWT [n][h] bf16
__global__ void cvt_t(const float* __restrict__ src, u16* __restrict__ dst) {
  int idx = blockIdx.x * blockDim.x + threadIdx.x;
  if (idx >= H_ * H_) return;
  int n = idx / H_, h = idx % H_;
  dst[idx] = f2bf(src[h * H_ + n]);
}

// ---------------- persistent LSTM kernel pieces ----------------
DEV void gbarrier(int* slots, int wg, int val, int tid) {
  __threadfence();
  __syncthreads();
  if (tid == 0) {
    __hip_atomic_store(slots + wg, val, __ATOMIC_RELEASE, __HIP_MEMORY_SCOPE_AGENT);
    for (int i = 0; i < NWG; ++i) {
      while (__hip_atomic_load(slots + i, __ATOMIC_ACQUIRE, __HIP_MEMORY_SCOPE_AGENT) < val) {}
    }
  }
  __syncthreads();
  __threadfence();
}

// bias added here (accumulators start at zero: zero-splat is rematerializable, never spilled)
DEV void gates_update(const v8f acc[4], const float* bv, float* cp, float* hrp) {
  #pragma unroll
  for (int r = 0; r < 8; ++r) {
    float ig = sigmoidf_(acc[0][r] + bv[0]);
    float fg = sigmoidf_(acc[1][r] + bv[1]);
    float gg = tanhf_(acc[2][r] + bv[2]);
    float og = sigmoidf_(acc[3][r] + bv[3]);
    float cn = fg * cp[r] + ig * gg;
    cp[r] = cn;
    hrp[r] = og * tanhf_(cn);
  }
}

// layer 0: g = x@Wih^T (global, K=32) + hprev@Whh^T (LDS, K=256) + b
DEV void layer0(const u16* __restrict__ xr, const u16* __restrict__ Wih,
                const u16* __restrict__ Whh, const u16* hprev,
                const float (*bv)[4], float (*c)[8], float (*hr)[8],
                int wave, int lane) {
  #pragma unroll
  for (int p = 0; p < PATCH; ++p) {
    const int tile = wave * PATCH + p;
    const int mt = tile >> 4, nt = tile & 15;
    v8f acc[4] = {{0, 0, 0, 0, 0, 0, 0, 0}, {0, 0, 0, 0, 0, 0, 0, 0},
                  {0, 0, 0, 0, 0, 0, 0, 0}, {0, 0, 0, 0, 0, 0, 0, 0}};
    {  // x panel (single K-chunk, weights zero-padded to K=32)
      v16bf A = load_a(xr + (size_t)(mt * 16) * 32, 32, lane);
      v16bf Bm[4];
      #pragma unroll
      for (int g = 0; g < 4; ++g)
        Bm[g] = load_b(Wih + (size_t)(g * H_ + nt * 16) * 32, 32, lane);
      #pragma unroll
      for (int g = 0; g < 4; ++g) acc[g] = wmma_bf(A, Bm[g], acc[g]);
    }
    #pragma unroll 2
    for (int kc = 0; kc < 8; ++kc) {  // recurrent panel from LDS
      v16bf A = load_a(hprev + (size_t)(mt * 16) * H_ + kc * 32, H_, lane);
      v16bf Bm[4];
      #pragma unroll
      for (int g = 0; g < 4; ++g)
        Bm[g] = load_b(Whh + (size_t)(g * H_ + nt * 16) * H_ + kc * 32, H_, lane);
      #pragma unroll
      for (int g = 0; g < 4; ++g) acc[g] = wmma_bf(A, Bm[g], acc[g]);
    }
    gates_update(acc, bv[p], c[p], hr[p]);
  }
}

// layer 1: g = h0@Wih^T + hprev@Whh^T + b (both activations in LDS, K=256 each)
DEV void layer1(const u16* h0s, const u16* __restrict__ Wih,
                const u16* __restrict__ Whh, const u16* hprev,
                const float (*bv)[4], float (*c)[8], float (*hr)[8],
                int wave, int lane) {
  #pragma unroll
  for (int p = 0; p < PATCH; ++p) {
    const int tile = wave * PATCH + p;
    const int mt = tile >> 4, nt = tile & 15;
    v8f acc[4] = {{0, 0, 0, 0, 0, 0, 0, 0}, {0, 0, 0, 0, 0, 0, 0, 0},
                  {0, 0, 0, 0, 0, 0, 0, 0}, {0, 0, 0, 0, 0, 0, 0, 0}};
    #pragma unroll 2
    for (int kc = 0; kc < 8; ++kc) {
      v16bf A = load_a(h0s + (size_t)(mt * 16) * H_ + kc * 32, H_, lane);
      v16bf Bm[4];
      #pragma unroll
      for (int g = 0; g < 4; ++g)
        Bm[g] = load_b(Wih + (size_t)(g * H_ + nt * 16) * H_ + kc * 32, H_, lane);
      #pragma unroll
      for (int g = 0; g < 4; ++g) acc[g] = wmma_bf(A, Bm[g], acc[g]);
    }
    #pragma unroll 2
    for (int kc = 0; kc < 8; ++kc) {
      v16bf A = load_a(hprev + (size_t)(mt * 16) * H_ + kc * 32, H_, lane);
      v16bf Bm[4];
      #pragma unroll
      for (int g = 0; g < 4; ++g)
        Bm[g] = load_b(Whh + (size_t)(g * H_ + nt * 16) * H_ + kc * 32, H_, lane);
      #pragma unroll
      for (int g = 0; g < 4; ++g) acc[g] = wmma_bf(A, Bm[g], acc[g]);
    }
    gates_update(acc, bv[p], c[p], hr[p]);
  }
}

DEV void write_h(u16* dst, const float (*hr)[8], int wave, int lane) {
  const int col = lane & 15, hi = lane >> 4;
  #pragma unroll
  for (int p = 0; p < PATCH; ++p) {
    const int tile = wave * PATCH + p;
    const int mt = tile >> 4, nt = tile & 15;
    #pragma unroll
    for (int r = 0; r < 8; ++r) {
      int m = mt * 16 + r + hi * 8;
      dst[(size_t)m * H_ + nt * 16 + col] = f2bf(hr[p][r]);
    }
  }
}

// hin = relu(h1 @ win[:,:256]^T + lev * win[:,256])
DEV void head_hin(const u16* h1s, const u16* __restrict__ win_s, const float* lev,
                  u16* hin, int wave, int lane) {
  const int col = lane & 15, hi = lane >> 4;
  #pragma unroll
  for (int p = 0; p < PATCH; ++p) {
    const int tile = wave * PATCH + p;
    const int mt = tile >> 4, nt = tile & 15;
    v8f acc = {0, 0, 0, 0, 0, 0, 0, 0};
    #pragma unroll 2
    for (int kc = 0; kc < 8; ++kc) {
      v16bf A  = load_a(h1s + (size_t)(mt * 16) * H_ + kc * 32, H_, lane);
      v16bf Bm = load_b(win_s + (size_t)(nt * 16) * 288 + kc * 32, 288, lane);
      acc = wmma_bf(A, Bm, acc);
    }
    float wl = bf2f(win_s[(size_t)(nt * 16 + col) * 288 + 256]);
    #pragma unroll
    for (int r = 0; r < 8; ++r) {
      int m = mt * 16 + r + hi * 8;
      float v = acc[r] + lev[m] * wl;
      hin[(size_t)m * H_ + nt * 16 + col] = f2bf(fmaxf(v, 0.0f));
    }
  }
}

// XW = hin @ gW  (gW stored transposed)
DEV void head_xw(const u16* hins, const u16* __restrict__ gwt, float* __restrict__ xw_site,
                 int ch, int wave, int lane) {
  const int col = lane & 15, hi = lane >> 4;
  #pragma unroll
  for (int p = 0; p < PATCH; ++p) {
    const int tile = wave * PATCH + p;
    const int mt = tile >> 4, nt = tile & 15;
    v8f acc = {0, 0, 0, 0, 0, 0, 0, 0};
    #pragma unroll 2
    for (int kc = 0; kc < 8; ++kc) {
      v16bf A  = load_a(hins + (size_t)(mt * 16) * H_ + kc * 32, H_, lane);
      v16bf Bm = load_b(gwt + (size_t)(nt * 16) * H_ + kc * 32, H_, lane);
      acc = wmma_bf(A, Bm, acc);
    }
    #pragma unroll
    for (int r = 0; r < 8; ++r) {
      int m = mt * 16 + r + hi * 8;
      xw_site[(size_t)(ch * CB + m) * H_ + nt * 16 + col] = acc[r];
    }
  }
}

__global__ void __launch_bounds__(THREADS, 1)
lstm_persistent(const float* __restrict__ enc_b0, const float* __restrict__ enc_b1,
                const float* __restrict__ df_b0,  const float* __restrict__ df_b1,
                const float* __restrict__ du_b0,  const float* __restrict__ du_b1,
                const float* __restrict__ level0, const float* __restrict__ flow_std,
                const float* __restrict__ adj,    const float* __restrict__ gb,
                const float* __restrict__ wout,
                char* __restrict__ ws, float* __restrict__ out) {
  __shared__ __align__(16) u16 h0[CB * H_];
  __shared__ __align__(16) u16 h1[CB * H_];
  __shared__ __align__(16) u16 hin[CB * H_];
  __shared__ float lev[CB];

  const int wg = blockIdx.x;
  const int s = wg / CHUNKS, ch = wg % CHUNKS;
  const int tid = threadIdx.x, wave = tid >> 5, lane = tid & 31;
  const int col = lane & 15;

  int* slots = (int*)(ws + B_BAR);
  const u16* WBase = (const u16*)(ws + B_W);
  const u16* win_s = (const u16*)(ws + B_WIN) + (size_t)s * H_ * 288;
  const u16* gwt   = (const u16*)(ws + B_GWT);
  float* XW = (float*)(ws + B_XW);

  float adjrow[S_];
  #pragma unroll
  for (int t = 0; t < S_; ++t) adjrow[t] = adj[s * S_ + t];
  const float fstd_s = flow_std[s];

  float c0[PATCH][8], c1[PATCH][8], hr[PATCH][8];
  #pragma unroll
  for (int p = 0; p < PATCH; ++p)
    #pragma unroll
    for (int r = 0; r < 8; ++r) { c0[p][r] = 0.0f; c1[p][r] = 0.0f; }

  for (int i = tid; i < CB * H_; i += THREADS) { h0[i] = 0; h1[i] = 0; }
  __syncthreads();

  int hs = 0;  // decoder step counter (barrier epoch / XW parity)
  for (int ph = 0; ph < 3; ++ph) {
    // per-phase parameters via integer selects (keeps pointers address-space clean)
    const size_t xoff = (ph == 0) ? B_XE : (ph == 1) ? B_XF : B_XU;
    const int    Td   = (ph == 0) ? T_  : (ph == 1) ? FH_  : FU_;
    const int    col0 = (ph == 2) ? FH_ : 0;
    const u16* Xp = (const u16*)(ws + xoff);
    const u16* P  = WBase + (size_t)ph * PFX_USH;
    const u16* WIH0 = P + O_WIH0 + (size_t)s * G4H * 32;
    const u16* WHH0 = P + O_WHH0 + (size_t)s * G4H * H_;
    const u16* WIH1 = P + O_WIH1 + (size_t)s * G4H * H_;
    const u16* WHH1 = P + O_WHH1 + (size_t)s * G4H * H_;
    const float* b0p = ((ph == 0) ? enc_b0 : (ph == 1) ? df_b0 : du_b0) + s * G4H;
    const float* b1p = ((ph == 0) ? enc_b1 : (ph == 1) ? df_b1 : du_b1) + s * G4H;

    // hoist bias values into scalar registers for the whole phase
    float bv0[PATCH][4], bv1[PATCH][4];
    #pragma unroll
    for (int p = 0; p < PATCH; ++p) {
      const int nt = (wave * PATCH + p) & 15;
      #pragma unroll
      for (int g = 0; g < 4; ++g) {
        bv0[p][g] = b0p[g * H_ + nt * 16 + col];
        bv1[p][g] = b1p[g * H_ + nt * 16 + col];
      }
    }

    for (int t = 0; t < Td; ++t) {
      const u16* xr = Xp + ((size_t)(s * Td + t) * Bt + ch * CB) * 32;
      layer0(xr, WIH0, WHH0, h0, bv0, c0, hr, wave, lane);
      __syncthreads();
      write_h(h0, hr, wave, lane);
      __syncthreads();
      layer1(h0, WIH1, WHH1, h1, bv1, c1, hr, wave, lane);
      __syncthreads();
      write_h(h1, hr, wave, lane);
      __syncthreads();

      if (ph > 0) {
        head_hin(h1, win_s, lev, hin, wave, lane);
        __syncthreads();
        const int par = hs & 1;
        float* xw_site = XW + ((size_t)par * S_ + s) * Bt * H_;
        head_xw(hin, gwt, xw_site, ch, wave, lane);
        gbarrier(slots, wg, hs + 1, tid);

        // graph mix + output head: one batch row per PARTS threads
        const float* xwp = XW + (size_t)par * S_ * Bt * H_;
        const int l = tid / PARTS, part = tid % PARTS;
        const int b = ch * CB + l;
        constexpr int NPER = H_ / PARTS;  // 16 columns per partial lane
        float partial = 0.0f;
        for (int n = part * NPER; n < part * NPER + NPER; ++n) {
          float gsum = gb[n];
          #pragma unroll
          for (int tt = 0; tt < S_; ++tt) {
            float a = adjrow[tt];
            if (a != 0.0f) gsum += a * xwp[((size_t)tt * Bt + b) * H_ + n];
          }
          gsum = fmaxf(gsum, 0.0f);
          partial += wout[s * H_ + n] * gsum;
        }
        partial += __shfl_xor(partial, 1, 32);
        partial += __shfl_xor(partial, 2, 32);
        partial += __shfl_xor(partial, 4, 32);
        partial += __shfl_xor(partial, 8, 32);
        if (part == 0) {
          const int colo = col0 + t;
          out[(size_t)b * (FH_ + FU_) * S_ + (size_t)colo * S_ + s] = partial;
          lev[l] += partial * fstd_s;
        }
        __syncthreads();
        ++hs;
      }
    }
    if (ph == 0) {  // initialize level before decoding starts
      if (tid < CB) lev[tid] = level0[(size_t)s * Bt + ch * CB + tid];
      __syncthreads();
    }
  }
}

// ---------------- host side ----------------
extern "C" void kernel_launch(void* const* d_in, const int* in_sizes, int n_in,
                              void* d_out, int out_size, void* d_ws, size_t ws_size,
                              hipStream_t stream) {
  (void)in_sizes; (void)n_in; (void)out_size; (void)ws_size;
  char* ws = (char*)d_ws;

  // barrier slots must start at zero every call (graph-capture-safe memset node)
  hipMemsetAsync(d_ws, 0, 4096, stream);

  auto blocks = [](size_t n) { return dim3((unsigned)((n + 255) / 256)); };
  const dim3 blk(256);

  // x tensors -> bf16 [S,T,B,32]
  u16* XE = (u16*)(ws + B_XE);
  u16* XF = (u16*)(ws + B_XF);
  u16* XU = (u16*)(ws + B_XU);
  cvt_x<<<blocks((size_t)S_ * T_  * Bt * 32), blk, 0, stream>>>((const float*)d_in[0], XE, T_, 16);
  cvt_x<<<blocks((size_t)S_ * FH_ * Bt * 32), blk, 0, stream>>>((const float*)d_in[1], XF, FH_, 16);
  cvt_x<<<blocks((size_t)S_ * FU_ * Bt * 32), blk, 0, stream>>>((const float*)d_in[2], XU, FU_, 4);

  // weights -> bf16 [N][K] rows (pad layer-0 input K to 32)
  const int in0[3] = {16, 16, 4};
  for (int p = 0; p < 3; ++p) {
    const int ib = 6 + p * 6;
    u16* P = (u16*)(ws + B_W) + (size_t)p * PFX_USH;
    cvt_pad<<<blocks((size_t)S_ * G4H * 32),  blk, 0, stream>>>((const float*)d_in[ib + 0], P + O_WIH0, S_ * G4H, in0[p], 32);
    cvt_pad<<<blocks((size_t)S_ * G4H * H_), blk, 0, stream>>>((const float*)d_in[ib + 1], P + O_WHH0, S_ * G4H, H_, H_);
    cvt_pad<<<blocks((size_t)S_ * G4H * H_), blk, 0, stream>>>((const float*)d_in[ib + 3], P + O_WIH1, S_ * G4H, H_, H_);
    cvt_pad<<<blocks((size_t)S_ * G4H * H_), blk, 0, stream>>>((const float*)d_in[ib + 4], P + O_WHH1, S_ * G4H, H_, H_);
  }
  // win [S,H,257] -> [S,H,288] bf16 ; gW -> gW^T bf16
  cvt_pad<<<blocks((size_t)S_ * H_ * 288), blk, 0, stream>>>((const float*)d_in[24], (u16*)(ws + B_WIN), S_ * H_, H_ + 1, 288);
  cvt_t  <<<blocks((size_t)H_ * H_),       blk, 0, stream>>>((const float*)d_in[26], (u16*)(ws + B_GWT));

  // one persistent kernel: whole encoder + both decoders
  lstm_persistent<<<dim3(NWG), dim3(THREADS), 0, stream>>>(
      (const float*)d_in[8],  (const float*)d_in[11],   // enc_b0, enc_b1
      (const float*)d_in[14], (const float*)d_in[17],   // df_b0,  df_b1
      (const float*)d_in[20], (const float*)d_in[23],   // du_b0,  du_b1
      (const float*)d_in[3],  (const float*)d_in[4],    // level0, flow_std
      (const float*)d_in[5],  (const float*)d_in[27],   // adj, gb
      (const float*)d_in[25],                           // wout
      ws, (float*)d_out);
}
